// VariationalAutoencoder_19456201851177
// MI455X (gfx1250) — matrix operand order repaired and checked by reading, not compiled, not run on hardware
//
#include <hip/hip_runtime.h>

// ---------------------------------------------------------------------------
// MI455X (gfx1250): wave32, WMMA 16x16x32 f16 (f32 accum) implicit-GEMM for
// all conv/fc layers; TDM (tensor_load_to_lds) stages weight tiles into LDS;
// latent covariance collapsed from 800^3 complex einsums to batched
// 100x100x100 real WMMA + 8x8 broadcast expansion (pure HBM-write bound).
// ---------------------------------------------------------------------------

typedef __attribute__((ext_vector_type(16))) _Float16 v16h;
typedef __attribute__((ext_vector_type(8)))  float    v8f;
typedef __attribute__((ext_vector_type(2)))  float    v2f;
typedef __attribute__((ext_vector_type(4)))  unsigned int u32x4;
typedef __attribute__((ext_vector_type(8)))  int      i32x8;
typedef __attribute__((ext_vector_type(4)))  int      i32x4;

#define BB   50
#define LLn  8
#define MMn  100
#define EIGn 10000
#define MLn  800

#if defined(__has_builtin)
#if __has_builtin(__builtin_amdgcn_tensor_load_to_lds) && __has_builtin(__builtin_amdgcn_s_wait_tensorcnt)
#define HAVE_TDM 1
#endif
#endif
#ifndef HAVE_TDM
#define HAVE_TDM 0
#endif

// ---------------- A-side loaders (implicit im2col, branch-free) ------------
struct APlain {
  const float* p; int K;
  static constexpr bool vec4 = true;
  __device__ const float* base(int m) const { return p + (long)m * K; }
  __device__ float ldc(int m, int k, int Ms, int Ks) const {
    bool ok = (m < Ms) & (k < Ks);
    long idx = ok ? ((long)m * K + k) : 0;
    float v = p[idx];
    return ok ? v : 0.f;
  }
};

struct AConv {  // forward conv 3x3, NCHW input
  const float* x; int Cin, H, W, Wo, HWo, stride, pad;
  static constexpr bool vec4 = false;
  __device__ const float* base(int) const { return x; }
  __device__ float ldc(int m, int k, int Ms, int Ks) const {
    int b = m / HWo; int rm = m - b * HWo;
    int ho = rm / Wo; int wo = rm - ho * Wo;
    int ci = k / 9; int rs = k - ci * 9; int r = rs / 3; int s = rs - r * 3;
    int hi = ho * stride - pad + r;
    int wi = wo * stride - pad + s;
    bool ok = (m < Ms) & (k < Ks) &
              ((unsigned)hi < (unsigned)H) & ((unsigned)wi < (unsigned)W);
    long idx = ok ? ((((long)b * Cin + ci) * H + hi) * W + wi) : 0;
    float v = x[idx];
    return ok ? v : 0.f;
  }
};

struct AConvT { // transposed-conv gather (PyTorch ConvTranspose2d semantics)
  const float* x; int Cin, H, W, Wo, HWo, stride, pad;
  static constexpr bool vec4 = false;
  __device__ const float* base(int) const { return x; }
  __device__ float ldc(int m, int k, int Ms, int Ks) const {
    int b = m / HWo; int rm = m - b * HWo;
    int ho = rm / Wo; int wo = rm - ho * Wo;
    int ci = k / 9; int rs = k - ci * 9; int r = rs / 3; int s = rs - r * 3;
    int th = ho + pad - r; int tw = wo + pad - s;
    int hi = th / stride,  wi = tw / stride;
    bool ok = (m < Ms) & (k < Ks) & (th >= 0) & (tw >= 0) &
              (hi * stride == th) & (wi * stride == tw) & (hi < H) & (wi < W);
    long idx = ok ? ((((long)b * Cin + ci) * H + hi) * W + wi) : 0;
    float v = x[idx];
    return ok ? v : 0.f;
  }
};

// ---------------- B-side loaders (element B[k][n], f32 source) -------------
struct BTrans {  // weight stored (N x K) row-major: OIHW conv / fc / lin
  const float* w; int K;
  static constexpr bool use_tdm = true;
  __device__ float ld(int k, int n, int Ks, int Ns) const {
    bool ok = (k < Ks) & (n < Ns);
    long idx = ok ? ((long)n * K + k) : 0;
    float v = w[idx];
    return ok ? v : 0.f;
  }
  __device__ unsigned long long tile_addr(int n, int k) const {
    return (unsigned long long)(size_t)(w + (long)n * K + k);
  }
};
struct BConvT {  // ConvTranspose weight IOHW: (Cin, Cout, 3, 3); k=(ci,rs)
  const float* w; int Cout; int K; // K unused (kept for interface parity)
  static constexpr bool use_tdm = false;
  __device__ float ld(int k, int n, int Ks, int Ns) const {
    int ci = k / 9; int rs = k - ci * 9;
    bool ok = (k < Ks) & (n < Ns);
    long idx = ok ? (((long)ci * Cout + n) * 9 + rs) : 0;
    float v = w[idx];
    return ok ? v : 0.f;
  }
  __device__ unsigned long long tile_addr(int, int) const { return 0; }
};

// ---------------- cooperative B-tile stage (fallback / BConvT) -------------
template <class BL>
__device__ inline void coop_stage(float* Bt, const BL& Bl, int k0, int n0,
                                  int Ks, int Ns) {
  for (int i = threadIdx.x; i < 64 * 32; i += 128) {
    int n = i >> 5, kk = i & 31;             // consecutive threads: consec k
    Bt[n * 33 + kk] = Bl.ld(k0 + kk, n0 + n, Ks, Ns);
  }
}

// ---------------- Generic WMMA GEMM ---------------------------------------
struct GP {
  int M, N, K;
  int act;    // 0 none, 1 leaky-relu(0.01), 2 relu
  int smode;  // 0 row-major (M x N), 1 NCHW scatter (m=(b,pos), n=channel)
  int HWo;    // spatial size for smode==1
  const float* bias;
  float* out;
};

// 128 threads = 4 waves; block tile = 64(M) x 64(N); k-step 32.
// Wave w owns m-tile w; loops 4 n-tiles reusing one A fragment (4 WMMA/step).
template <class AL, class BL>
__global__ __launch_bounds__(128) void gemm16(AL A, BL Bl, GP p) {
  __shared__ float Bt[64 * 33];              // 33-float pitch: bank-conflict free
  const int lane = threadIdx.x & 31;
  const int wave = threadIdx.x >> 5;
  const int n0   = blockIdx.x * 64;
  const int m0   = (blockIdx.y * 4 + wave) * 16;
  const int mA   = m0 + (lane & 15);         // A-fragment row for this lane
  const int kb   = (lane >> 4) * 8;          // A-fragment K sub-base
  const int ksB  = (lane >> 4) * 16;         // unused; B fragment reads use kb
  const int nB   = lane & 15;                // B/C fragment column
  const bool inM = (mA < p.M);
  v8f acc[4] = {};

  for (int k0 = 0; k0 < p.K; k0 += 32) {
    __syncthreads();                         // previous fragment reads done
#if HAVE_TDM
    if constexpr (BL::use_tdm) {
      // each wave DMAs its 16(n) x 32(k) f32 slice; TDM pad inserts 1 dword
      // per 32 -> LDS pitch 33 to match fragment reads.
      const int w    = __builtin_amdgcn_readfirstlane(wave);
      const int nrow = n0 + w * 16;
      unsigned ldsb  = (unsigned)(size_t)(&Bt[0]) + (unsigned)(w * 16 * 33 * 4);
      unsigned long long ga = Bl.tile_addr(nrow, k0);
      int rem0 = p.K - k0;                         // remaining k extent
      int rem1 = p.N - nrow; if (rem1 < 0) rem1 = 0; // remaining n rows
      u32x4 g0;
      g0[0] = 1u;                                  // count=1, user descriptor
      g0[1] = ldsb;                                // lds_addr
      g0[2] = (unsigned)(ga & 0xffffffffull);      // global_addr[31:0] @bit64
      g0[3] = (unsigned)((ga >> 32) & 0x01ffffffull) | (2u << 30); // type=2
      i32x8 g1;
      g1[0] = (2 << 16) | (1 << 20) | (4 << 22);   // 4B elems, pad_en, 32dw int.
      g1[1] = (rem0 & 0xffff) << 16;               // tensor_dim0[15:0]
      g1[2] = ((rem0 >> 16) & 0xffff) | ((rem1 & 0xffff) << 16);
      g1[3] = ((rem1 >> 16) & 0xffff) | (32 << 16);// tile_dim0 = 32
      g1[4] = 16;                                  // tile_dim1 = 16
      g1[5] = Bl.K;                                // dim0 stride (elements)
      g1[6] = 0; g1[7] = 0;
      i32x4 z4 = {};
#if __clang_major__ >= 23
      i32x8 z8 = {};
      __builtin_amdgcn_tensor_load_to_lds(g0, g1, z4, z4, z8, 0);
#else
      __builtin_amdgcn_tensor_load_to_lds(g0, g1, z4, z4, 0);
#endif
      __builtin_amdgcn_s_wait_tensorcnt(0);
    } else {
      coop_stage(Bt, Bl, k0, n0, p.K, p.N);
    }
#else
    coop_stage(Bt, Bl, k0, n0, p.K, p.N);
#endif
    __syncthreads();

    // ---- A fragment (ISA layout: K = kb..kb+7 and kb+16..kb+23) ----
    v16h a;
    bool fast = false;
    if constexpr (AL::vec4) {
      if (inM && (k0 + 32 <= p.K)) {
        const float* bse = A.base(mA) + k0 + kb;
        float4 f0 = *(const float4*)(bse);
        float4 f1 = *(const float4*)(bse + 4);
        float4 f2 = *(const float4*)(bse + 16);
        float4 f3 = *(const float4*)(bse + 20);
        a[0]=(_Float16)f0.x; a[1]=(_Float16)f0.y; a[2]=(_Float16)f0.z; a[3]=(_Float16)f0.w;
        a[4]=(_Float16)f1.x; a[5]=(_Float16)f1.y; a[6]=(_Float16)f1.z; a[7]=(_Float16)f1.w;
        a[8]=(_Float16)f2.x; a[9]=(_Float16)f2.y; a[10]=(_Float16)f2.z; a[11]=(_Float16)f2.w;
        a[12]=(_Float16)f3.x; a[13]=(_Float16)f3.y; a[14]=(_Float16)f3.z; a[15]=(_Float16)f3.w;
        fast = true;
        if (k0 + 32 < p.K) __builtin_prefetch(bse + 32, 0, 3);
      }
    }
    if (!fast) {
#pragma unroll
      for (int j = 0; j < 8; ++j) {
        a[j]     = (_Float16)A.ldc(mA, k0 + kb + j,      p.M, p.K);
        a[j + 8] = (_Float16)A.ldc(mA, k0 + kb + 16 + j, p.M, p.K);
      }
    }

    // ---- 4 n-tiles per wave, reuse A fragment ----
#pragma unroll
    for (int t = 0; t < 4; ++t) {
      v16h b;
#pragma unroll
      for (int j = 0; j < 16; ++j)
        b[j] = (_Float16)Bt[(t * 16 + nB) * 33 + kb + (j & 7) + ((j >> 3) << 4)];
      acc[t] = __builtin_amdgcn_wmma_f32_16x16x32_f16(
          false, a, false, b, (short)0, acc[t], false, false);
    }
  }

  // ---- epilogue: lane holds rows m0 + v + 8*(lane>>4), col n0+t*16+nB ----
#pragma unroll
  for (int t = 0; t < 4; ++t) {
    int n = n0 + t * 16 + nB;
    if (n >= p.N) continue;
    float bv = p.bias ? p.bias[n] : 0.f;
#pragma unroll
    for (int v = 0; v < 8; ++v) {
      int m = m0 + (lane >> 4) * 8 + v;
      if (m >= p.M) continue;
      float val = acc[t][v] + bv;
      if (p.act == 1)      val = (val > 0.f) ? val : 0.01f * val;
      else if (p.act == 2) val = fmaxf(val, 0.f);
      long idx;
      if (p.smode == 0) idx = (long)m * p.N + n;
      else {
        int bI = m / p.HWo; int pos = m - bI * p.HWo;
        idx = ((long)bI * p.N + n) * p.HWo + pos;
      }
      p.out[idx] = val;
    }
  }
}

// ---------------- BatchNorm (training-mode batch stats) --------------------
__global__ void bn_stats(const float* x, float* mu, float* rstd,
                         int C, int HW, int Bn) {
  int c = blockIdx.x;
  __shared__ float s1[256], s2[256];
  float a = 0.f, b = 0.f;
  int N = Bn * HW;
  for (int i = threadIdx.x; i < N; i += 256) {
    int bb = i / HW, pos = i - bb * HW;
    float v = x[((long)bb * C + c) * HW + pos];
    a += v; b += v * v;
  }
  s1[threadIdx.x] = a; s2[threadIdx.x] = b;
  __syncthreads();
  for (int st = 128; st > 0; st >>= 1) {
    if (threadIdx.x < st) {
      s1[threadIdx.x] += s1[threadIdx.x + st];
      s2[threadIdx.x] += s2[threadIdx.x + st];
    }
    __syncthreads();
  }
  if (threadIdx.x == 0) {
    float m = s1[0] / (float)N;
    float var = s2[0] / (float)N - m * m;
    mu[c] = m; rstd[c] = rsqrtf(var + 1e-5f);
  }
}

__global__ void bn_apply(float* x, const float* mu, const float* rstd,
                         int C, int HW, long total) {
  long i = (long)blockIdx.x * blockDim.x + threadIdx.x;
  if (i < total) {
    int c = (int)((i / HW) % C);
    x[i] = (x[i] - mu[c]) * rstd[c];
  }
}

// ---------------- DFT base W[j,k] = exp(-2*pi*i * (j^k mod 100)/100) -------
__global__ void build_W(float* Wre, float* Wim) {
  int j = blockIdx.x, k = threadIdx.x;
  if (k >= MMn) return;
  int e = 1;                                  // pow(j,0,100)==1 (incl. j==0)
  for (int t = 0; t < k; ++t) e = (e * j) % MMn;
  double th = -2.0 * 3.14159265358979323846 * (double)e / (double)MMn;
  Wre[j * MMn + k] = (float)cos(th);
  Wim[j * MMn + k] = (float)sin(th);
}

// ---------------- C = FB Sigma FBH collapsed to 100x100 + 8x8 broadcast ----
#if __has_builtin(__builtin_amdgcn_wmma_f32_16x16x4_f32)
__global__ __launch_bounds__(32) void latent_C(const float* Wre, const float* Wim,
                                               const float* eig, float* C) {
  const int b = blockIdx.z;
  const int j0 = blockIdx.y * 16, k0 = blockIdx.x * 16;
  const int lane = threadIdx.x & 31;
  const int row = j0 + (lane & 15);
  const int col = k0 + (lane & 15);
  const int rowc = (row < MMn) ? row : (MMn - 1);   // clamp+select, no branch
  const int colc = (col < MMn) ? col : (MMn - 1);
  const bool okR = (row < MMn), okC = (col < MMn);
  v8f acc = {};
  for (int r0 = 0; r0 < MMn; r0 += 4) {
    const int rA = r0 + 2 * (lane >> 4);            // f32 frag K placement
    v2f are, aim, bre, bim;
#pragma unroll
    for (int t = 0; t < 2; ++t) {
      int r = rA + t;                               // always < 100
      float e  = eig[(long)b * EIGn + r];
      float wr = Wre[rowc * MMn + r], wi = Wim[rowc * MMn + r];
      float vr = Wre[colc * MMn + r], vi = Wim[colc * MMn + r];
      are[t] = okR ? wr * e : 0.f;
      aim[t] = okR ? wi * e : 0.f;
      bre[t] = okC ? vr : 0.f;
      bim[t] = okC ? vi : 0.f;
    }
    acc = __builtin_amdgcn_wmma_f32_16x16x4_f32(
        false, are, false, bre, (short)0, acc, false, false);
    acc = __builtin_amdgcn_wmma_f32_16x16x4_f32(
        false, aim, false, bim, (short)0, acc, false, false);
  }
  if (okC) {
#pragma unroll
    for (int v = 0; v < 8; ++v) {
      int j = j0 + (lane >> 4) * 8 + v;
      if (j >= MMn) continue;
      float val = acc[v] * 1e-4f;
      float2 pv = make_float2(val, val);
#pragma unroll
      for (int a = 0; a < 8; ++a) {     // broadcast 8x8 constant block
        float2* q = (float2*)(C + (long)b * 640000 + (long)(j * 8 + a) * 800 + col * 8);
        q[0] = pv; q[1] = pv; q[2] = pv; q[3] = pv;
      }
    }
  }
}
#else
__global__ void latent_C(const float* Wre, const float* Wim,
                         const float* eig, float* C) {
  int idx = blockIdx.x * blockDim.x + threadIdx.x;
  int b = blockIdx.y;
  if (idx >= MMn * MMn) return;
  int j = idx / MMn, k = idx - j * MMn;
  float s = 0.f;
  for (int r = 0; r < MMn; ++r)
    s += (Wre[j * MMn + r] * Wre[k * MMn + r] +
          Wim[j * MMn + r] * Wim[k * MMn + r]) * eig[(long)b * EIGn + r];
  s *= 1e-4f;
  float2 pv = make_float2(s, s);
  for (int a = 0; a < 8; ++a) {
    float2* q = (float2*)(C + (long)b * 640000 + (long)(j * 8 + a) * 800 + k * 8);
    q[0] = pv; q[1] = pv; q[2] = pv; q[3] = pv;
  }
}
#endif

// ---------------- z = Re(FBH (Sigma^2 zeta)) collapsed to 100-dim ----------
__global__ void latent_z(const float* Wre, const float* Wim, const float* eig,
                         const float* zre, const float* zim, float* zf) {
  int b = blockIdx.x;
  __shared__ float Sre[MMn], Sim[MMn];
  for (int r = threadIdx.x; r < MMn; r += blockDim.x) {
    float sr = 0.f, si = 0.f;
    for (int c = 0; c < LLn; ++c) {
      sr += zre[(long)b * MLn + r * LLn + c];
      si += zim[(long)b * MLn + r * LLn + c];
    }
    Sre[r] = sr; Sim[r] = si;
  }
  __syncthreads();
  for (int j = threadIdx.x; j < MMn; j += blockDim.x) {
    float acc = 0.f;
    for (int r = 0; r < MMn; ++r) {
      float e = eig[(long)b * EIGn + r];
      acc += e * e * (Wre[r * MMn + j] * Sre[r] + Wim[r * MMn + j] * Sim[r]);
    }
    acc *= 0.1f;
    for (int a = 0; a < LLn; ++a) zf[(long)b * MLn + j * LLn + a] = acc;
  }
}

// ---------------------------------------------------------------------------
extern "C" void kernel_launch(void* const* d_in, const int* in_sizes, int n_in,
                              void* d_out, int out_size, void* d_ws, size_t ws_size,
                              hipStream_t stream) {
  (void)in_sizes; (void)n_in; (void)out_size; (void)ws_size;
  const float* x    = (const float*)d_in[0];
  const float* zre  = (const float*)d_in[1];
  const float* zim  = (const float*)d_in[2];
  const float* w_c1 = (const float*)d_in[3];  const float* b_c1 = (const float*)d_in[4];
  const float* w_c2 = (const float*)d_in[5];  const float* b_c2 = (const float*)d_in[6];
  const float* w_c3 = (const float*)d_in[7];  const float* b_c3 = (const float*)d_in[8];
  const float* w_c4 = (const float*)d_in[9];  const float* b_c4 = (const float*)d_in[10];
  const float* w_fc = (const float*)d_in[11]; const float* b_fc = (const float*)d_in[12];
  const float* w_ln = (const float*)d_in[13]; const float* b_ln = (const float*)d_in[14];
  const float* w_t1 = (const float*)d_in[15]; const float* b_t1 = (const float*)d_in[16];
  const float* w_t2 = (const float*)d_in[17]; const float* b_t2 = (const float*)d_in[18];
  const float* w_t3 = (const float*)d_in[19]; const float* b_t3 = (const float*)d_in[20];
  const float* w_t4 = (const float*)d_in[21]; const float* b_t4 = (const float*)d_in[22];
  const float* w_t5 = (const float*)d_in[23]; const float* b_t5 = (const float*)d_in[24];

  float* out  = (float*)d_out;
  float* outD = out;                         // (50,1,49,49)  = 120050
  float* outC = out + 120050;                // (50,800,800)  = 32,000,000
  float* outE = out + 120050 + 32000000;     // (50,10000)    = 500,000

  float* ws    = (float*)d_ws;
  float* Wre   = ws;                         // 10000
  float* Wim   = ws + 10000;                 // 10000
  float* mu    = ws + 20000;                 // 1024
  float* rstd  = ws + 21536;                 // 1024
  float* zfull = ws + 23072;                 // 40000
  float* bufA  = ws + 65536;                 // 8M floats
  float* bufB  = bufA + (8 << 20);           // 8M floats

  auto gg = [](int M, int N) { return dim3((unsigned)((N + 63) / 64),
                                           (unsigned)((M + 63) / 64)); };

  // ----- encoder: conv -> lrelu (fused) -> batchnorm -----
  { AConv A{x, 1, 64, 64, 32, 1024, 2, 1};
    GP p{51200, 128, 9, 1, 1, 1024, b_c1, bufA};
    gemm16<<<gg(p.M, p.N), 128, 0, stream>>>(A, BTrans{w_c1, 9}, p);
    bn_stats<<<128, 256, 0, stream>>>(bufA, mu, rstd, 128, 1024, BB);
    bn_apply<<<(6553600 + 255) / 256, 256, 0, stream>>>(bufA, mu, rstd, 128, 1024, 6553600L); }

  { AConv A{bufA, 128, 32, 32, 16, 256, 2, 1};
    GP p{12800, 256, 1152, 1, 1, 256, b_c2, bufB};
    gemm16<<<gg(p.M, p.N), 128, 0, stream>>>(A, BTrans{w_c2, 1152}, p);
    bn_stats<<<256, 256, 0, stream>>>(bufB, mu, rstd, 256, 256, BB);
    bn_apply<<<(3276800 + 255) / 256, 256, 0, stream>>>(bufB, mu, rstd, 256, 256, 3276800L); }

  { AConv A{bufB, 256, 16, 16, 8, 64, 2, 1};
    GP p{3200, 1024, 2304, 1, 1, 64, b_c3, bufA};
    gemm16<<<gg(p.M, p.N), 128, 0, stream>>>(A, BTrans{w_c3, 2304}, p);
    bn_stats<<<1024, 256, 0, stream>>>(bufA, mu, rstd, 1024, 64, BB);
    bn_apply<<<(3276800 + 255) / 256, 256, 0, stream>>>(bufA, mu, rstd, 1024, 64, 3276800L); }

  { AConv A{bufA, 1024, 8, 8, 4, 16, 2, 1};
    GP p{800, 1024, 9216, 1, 1, 16, b_c4, bufB};
    gemm16<<<gg(p.M, p.N), 128, 0, stream>>>(A, BTrans{w_c4, 9216}, p);
    bn_stats<<<1024, 256, 0, stream>>>(bufB, mu, rstd, 1024, 16, BB);
    bn_apply<<<(819200 + 255) / 256, 256, 0, stream>>>(bufB, mu, rstd, 1024, 16, 819200L); }

  // ----- FC: eig = relu(hflat @ w_fc^T + b_fc) -> straight into d_out -----
  { GP p{50, 10000, 16384, 2, 0, 1, b_fc, outE};
    gemm16<<<gg(p.M, p.N), 128, 0, stream>>>(APlain{bufB, 16384}, BTrans{w_fc, 16384}, p); }

  // ----- latent: W matrix, covariance C, sample z -----
  build_W<<<100, 128, 0, stream>>>(Wre, Wim);
#if __has_builtin(__builtin_amdgcn_wmma_f32_16x16x4_f32)
  latent_C<<<dim3(7, 7, 50), 32, 0, stream>>>(Wre, Wim, outE, outC);
#else
  latent_C<<<dim3((MMn * MMn + 255) / 256, 50), 256, 0, stream>>>(Wre, Wim, outE, outC);
#endif
  latent_z<<<50, 128, 0, stream>>>(Wre, Wim, outE, zre, zim, zfull);

  // ----- decoder linear: (50,800) @ (800,9216) -> (50,1024,3,3) -----
  { GP p{50, 9216, 800, 0, 0, 1, b_ln, bufA};
    gemm16<<<gg(p.M, p.N), 128, 0, stream>>>(APlain{zfull, 800}, BTrans{w_ln, 800}, p); }

  // ----- decoder transposed convs -----
  { AConvT A{bufA, 1024, 3, 3, 5, 25, 2, 1};
    GP p{1250, 512, 9216, 0, 1, 25, b_t1, bufB};
    gemm16<<<gg(p.M, p.N), 128, 0, stream>>>(A, BConvT{w_t1, 512, 0}, p); }

  { AConvT A{bufB, 512, 5, 5, 11, 121, 2, 0};
    GP p{6050, 256, 4608, 0, 1, 121, b_t2, bufA};
    gemm16<<<gg(p.M, p.N), 128, 0, stream>>>(A, BConvT{w_t2, 256, 0}, p); }

  { AConvT A{bufA, 256, 11, 11, 23, 529, 2, 0};
    GP p{26450, 128, 2304, 0, 1, 529, b_t3, bufB};
    gemm16<<<gg(p.M, p.N), 128, 0, stream>>>(A, BConvT{w_t3, 128, 0}, p); }

  { AConvT A{bufB, 128, 23, 23, 47, 2209, 2, 0};
    GP p{110450, 64, 1152, 0, 1, 2209, b_t4, bufA};
    gemm16<<<gg(p.M, p.N), 128, 0, stream>>>(A, BConvT{w_t4, 64, 0}, p); }

  { AConvT A{bufA, 64, 47, 47, 49, 2401, 1, 0};
    GP p{120050, 1, 576, 0, 1, 2401, b_t5, outD};
    gemm16<<<gg(p.M, p.N), 128, 0, stream>>>(A, BConvT{w_t5, 1, 0}, p); }
}
